// Flow_Attention_90237262889581
// MI455X (gfx1250) — compile-verified, matrix operand order checked by reading, not run on hardware
//
#include <hip/hip_runtime.h>
#include <hip/hip_bf16.h>

// ---------------------------------------------------------------------------
// Flow Attention for MI455X (gfx1250, wave32, WMMA bf16 + async/TDM staging)
// B=4, N=8192, C=512, H=8, D=64
// ---------------------------------------------------------------------------

typedef __attribute__((ext_vector_type(16))) __bf16 v16bf;
typedef __attribute__((ext_vector_type(8)))  float  v8f;
typedef __attribute__((ext_vector_type(4)))  float  v4f;
typedef __attribute__((ext_vector_type(4)))  unsigned int v4u;
typedef __attribute__((ext_vector_type(8)))  int    v8i;
typedef __attribute__((ext_vector_type(4)))  int    v4i;

#define FA_B 4
#define FA_N 8192
#define FA_C 512
#define FA_H 8
#define FA_D 64
#define FA_BH (FA_B * FA_H)     // 32
#define FA_EPS 1e-6f

__device__ __forceinline__ float sigm(float x) { return 1.0f / (1.0f + __expf(-x)); }

__device__ __forceinline__ v8f zero8() {
  v8f z;
#pragma unroll
  for (int i = 0; i < 8; ++i) z[i] = 0.0f;
  return z;
}

// Load a 16x32 bf16 WMMA A/B fragment from LDS.
// ISA layout (CDNA5 ISA, 16-bit A 16x32): lanes 0-15 hold K=0..7 (V0-3) and
// K=16..23 (V4-7); lanes 16-31 hold K=8..15 and K=24..31. Caller passes the
// row base pointer (row = tile_row*16 + (lane&15)); B is staged transposed so
// the same loader serves both operands.
__device__ __forceinline__ v16bf load_frag(const __bf16* p, int lane) {
  const int kb = (lane >> 4) * 8;
  v16bf f;
#pragma unroll
  for (int i = 0; i < 8; ++i) {
    f[i]     = p[kb + i];
    f[i + 8] = p[16 + kb + i];
  }
  return f;
}

__device__ __forceinline__ v8f wmma_bf16(v16bf a, v16bf b, v8f c) {
  return __builtin_amdgcn_wmma_f32_16x16x32_bf16(false, a, false, b, (short)0, c,
                                                 false, false);
}

// ---------------------------------------------------------------------------
// Kernel 1: qkv = x @ W_qkv^T  (M=32768, N=1536, K=512), fused epilogue:
//   q = sigmoid, k = sigmoid, v = v/8, scatter to [B,H,N,D].
// Block tile 64x64, 256 threads (8 waves x 2 tiles).
// ---------------------------------------------------------------------------
__global__ __launch_bounds__(256) void qkv_wmma_kernel(
    const float* __restrict__ x, const float* __restrict__ Wqkv,
    float* __restrict__ q, float* __restrict__ k, float* __restrict__ v) {
  __shared__ __bf16 As[64][40];
  __shared__ __bf16 Bs[64][40];
  const int tid = threadIdx.x;
  const int lane = tid & 31, w = tid >> 5;
  const int lr = lane & 15;
  const int mt = w & 3, ntb = (w >> 2) * 2;
  const int m0 = blockIdx.x * 64;
  const int c0 = blockIdx.y * 64;

  v8f acc[2];
  acc[0] = zero8();
  acc[1] = zero8();

  for (int kk0 = 0; kk0 < FA_C; kk0 += 32) {
#pragma unroll
    for (int i = 0; i < 2; ++i) {
      const int f = tid + i * 256;      // 512 float4 ids per operand
      const int row = f >> 3;           // 0..63
      const int c4 = (f & 7) * 4;       // 0,4,..28
      v4f xv = *(const v4f*)(x + (size_t)(m0 + row) * FA_C + kk0 + c4);
      v4f wv = *(const v4f*)(Wqkv + (size_t)(c0 + row) * FA_C + kk0 + c4);
#pragma unroll
      for (int j = 0; j < 4; ++j) {
        As[row][c4 + j] = (__bf16)xv[j];
        Bs[row][c4 + j] = (__bf16)wv[j];
      }
      if (kk0 + 32 < FA_C)
        __builtin_prefetch(x + (size_t)(m0 + row) * FA_C + kk0 + 32 + c4, 0, 1);
    }
    __syncthreads();
    v16bf af = load_frag(&As[mt * 16 + lr][0], lane);
#pragma unroll
    for (int t = 0; t < 2; ++t) {
      v16bf bf = load_frag(&Bs[(ntb + t) * 16 + lr][0], lane);
      acc[t] = wmma_bf16(af, bf, acc[t]);
    }
    __syncthreads();
  }

#pragma unroll
  for (int t = 0; t < 2; ++t) {
#pragma unroll
    for (int r = 0; r < 8; ++r) {
      const int gm = m0 + mt * 16 + (lane >> 4) * 8 + r;  // global row (b*N+n)
      const int gc = c0 + (ntb + t) * 16 + lr;            // [0,1536)
      const int which = gc >> 9;     // 0=q 1=k 2=v
      const int ci = gc & 511;
      const int h = ci >> 6, d = ci & 63;
      const int b = gm >> 13, n = gm & 8191;
      const size_t idx = (((size_t)(b * FA_H + h) * FA_N) + n) * FA_D + d;
      const float val = acc[t][r];
      if (which == 0)      q[idx] = sigm(val);
      else if (which == 1) k[idx] = sigm(val);
      else                 v[idx] = val * 0.125f;   // v / ((D^-0.5 * C)//H) = v/8
    }
  }
}

// ---------------------------------------------------------------------------
// Kernel 2: column sums over N -> q_sum/k_sum [BH, D]
// ---------------------------------------------------------------------------
__global__ __launch_bounds__(256) void colsum_kernel(
    const float* __restrict__ q, const float* __restrict__ k,
    float* __restrict__ qsum, float* __restrict__ ksum) {
  const int bh = blockIdx.x;
  const int d = threadIdx.x & 63, sl = threadIdx.x >> 6;
  const float* qb = q + (size_t)bh * FA_N * FA_D;
  const float* kb = k + (size_t)bh * FA_N * FA_D;
  float sq = 0.0f, sk = 0.0f;
  for (int n = sl; n < FA_N; n += 4) {
    sq += qb[(size_t)n * FA_D + d];
    sk += kb[(size_t)n * FA_D + d];
  }
  __shared__ float lq[4][64], lk[4][64];
  lq[sl][d] = sq;
  lk[sl][d] = sk;
  __syncthreads();
  if (sl == 0) {
    qsum[bh * 64 + d] = lq[0][d] + lq[1][d] + lq[2][d] + lq[3][d];
    ksum[bh * 64 + d] = lk[0][d] + lk[1][d] + lk[2][d] + lk[3][d];
  }
}

// ---------------------------------------------------------------------------
// Kernel 3: per-token sink_incoming/source_outgoing; accumulate
//   ks_so[d] = sum_n k*so,  qs_si[d] = sum_n q*si  (LDS partial + global atomics)
// ---------------------------------------------------------------------------
__global__ __launch_bounds__(256) void flow1_kernel(
    const float* __restrict__ q, const float* __restrict__ k,
    const float* __restrict__ qsum, const float* __restrict__ ksum,
    float* __restrict__ si, float* __restrict__ so,
    float* __restrict__ ks_so, float* __restrict__ qs_si) {
  const int bh = blockIdx.y;
  const int n = blockIdx.x * 256 + threadIdx.x;
  __shared__ float lks[64], lqs[64], pk[64], pq[64];
  if (threadIdx.x < 64) {
    lks[threadIdx.x] = ksum[bh * 64 + threadIdx.x] + FA_EPS;
    lqs[threadIdx.x] = qsum[bh * 64 + threadIdx.x] + FA_EPS;
    pk[threadIdx.x] = 0.0f;
    pq[threadIdx.x] = 0.0f;
  }
  __syncthreads();
  const size_t base = ((size_t)bh * FA_N + n) * FA_D;
  float dq = 0.0f, dk = 0.0f;
  for (int d = 0; d < FA_D; d += 4) {
    v4f qv = *(const v4f*)(q + base + d);
    v4f kv = *(const v4f*)(k + base + d);
#pragma unroll
    for (int j = 0; j < 4; ++j) {
      dq += (qv[j] + FA_EPS) * lks[d + j];
      dk += (kv[j] + FA_EPS) * lqs[d + j];
    }
  }
  const float si_ = 1.0f / (dq + FA_EPS);
  const float so_ = 1.0f / (dk + FA_EPS);
  si[(size_t)bh * FA_N + n] = si_;
  so[(size_t)bh * FA_N + n] = so_;
  for (int d = 0; d < FA_D; d += 4) {
    v4f qv = *(const v4f*)(q + base + d);
    v4f kv = *(const v4f*)(k + base + d);
#pragma unroll
    for (int j = 0; j < 4; ++j) {
      atomicAdd(&pk[d + j], kv[j] * so_);
      atomicAdd(&pq[d + j], qv[j] * si_);
    }
  }
  __syncthreads();
  if (threadIdx.x < 64) {
    atomicAdd(&ks_so[bh * 64 + threadIdx.x], pk[threadIdx.x]);
    atomicAdd(&qs_si[bh * 64 + threadIdx.x], pq[threadIdx.x]);
  }
}

// ---------------------------------------------------------------------------
// Kernel 4: conserved sink/source, sink allocation, softmax numerator + denom.
//   mult[n]  = si[n] * sigmoid(conserved_sink)    (combined epilogue scale)
//   expcs[n] = exp(clip(conserved_source,-1,1)),  denom[bh] += sum(expcs)
// ---------------------------------------------------------------------------
__global__ __launch_bounds__(256) void flow2_kernel(
    const float* __restrict__ q, const float* __restrict__ k,
    const float* __restrict__ ks_so, const float* __restrict__ qs_si,
    const float* __restrict__ si, float* __restrict__ mult,
    float* __restrict__ expcs, float* __restrict__ denom) {
  const int bh = blockIdx.y;
  const int n = blockIdx.x * 256 + threadIdx.x;
  __shared__ float lko[64], lqi[64];
  __shared__ float red[256];
  if (threadIdx.x < 64) {
    lko[threadIdx.x] = ks_so[bh * 64 + threadIdx.x] + FA_EPS;
    lqi[threadIdx.x] = qs_si[bh * 64 + threadIdx.x] + FA_EPS;
  }
  __syncthreads();
  const size_t base = ((size_t)bh * FA_N + n) * FA_D;
  float csk = 0.0f, css = 0.0f;
  for (int d = 0; d < FA_D; d += 4) {
    v4f qv = *(const v4f*)(q + base + d);
    v4f kv = *(const v4f*)(k + base + d);
#pragma unroll
    for (int j = 0; j < 4; ++j) {
      csk += (qv[j] + FA_EPS) * lko[d + j];
      css += (kv[j] + FA_EPS) * lqi[d + j];
    }
  }
  csk += FA_EPS;
  css += FA_EPS;
  css = fminf(1.0f, fmaxf(-1.0f, css));
  const float alloc = sigm(csk);  // * (N_q/N_k) == 1
  mult[(size_t)bh * FA_N + n] = si[(size_t)bh * FA_N + n] * alloc;
  const float e = __expf(css);
  expcs[(size_t)bh * FA_N + n] = e;
  red[threadIdx.x] = e;
  __syncthreads();
  for (int s = 128; s > 0; s >>= 1) {
    if (threadIdx.x < s) red[threadIdx.x] += red[threadIdx.x + s];
    __syncthreads();
  }
  if (threadIdx.x == 0) atomicAdd(&denom[bh], red[0]);
}

// ---------------------------------------------------------------------------
// Kernel 5: kv = k^T @ (v * source_competition)  per (b,h): 64x64, K=N=8192.
// Output stored transposed in bf16 (kvbT[e][d]) for direct B-fragment staging.
// ---------------------------------------------------------------------------
__global__ __launch_bounds__(256) void kv_wmma_kernel(
    const float* __restrict__ k, const float* __restrict__ v,
    const float* __restrict__ expcs, const float* __restrict__ denom,
    __bf16* __restrict__ kvbT) {
  const int bh = blockIdx.x;
  __shared__ __bf16 Ks[64][40];  // [m=d][kk]  (k transposed)
  __shared__ __bf16 Vs[64][40];  // [e][kk]    (v*sc transposed)
  const int tid = threadIdx.x;
  const int lane = tid & 31, w = tid >> 5;
  const int lr = lane & 15;
  const int mt = w & 3, ntb = (w >> 2) * 2;
  const float scN = (float)FA_N / denom[bh];
  const size_t kb = (size_t)bh * FA_N * FA_D;

  v8f acc[2];
  acc[0] = zero8();
  acc[1] = zero8();

  for (int n0 = 0; n0 < FA_N; n0 += 32) {
#pragma unroll
    for (int i = 0; i < 2; ++i) {
      const int g = tid + i * 256;    // 512 float4 ids per operand
      const int kk = g >> 4;          // 0..31
      const int c4 = (g & 15) * 4;    // 0..60
      const size_t off = kb + (size_t)(n0 + kk) * FA_D + c4;
      v4f kvec = *(const v4f*)(k + off);
      v4f vvec = *(const v4f*)(v + off);
      const float s = expcs[(size_t)bh * FA_N + n0 + kk] * scN;
#pragma unroll
      for (int j = 0; j < 4; ++j) {
        Ks[c4 + j][kk] = (__bf16)kvec[j];
        Vs[c4 + j][kk] = (__bf16)(vvec[j] * s);
      }
      if (n0 + 32 < FA_N) __builtin_prefetch(k + off + 32 * FA_D, 0, 1);
    }
    __syncthreads();
    v16bf af = load_frag(&Ks[mt * 16 + lr][0], lane);
#pragma unroll
    for (int t = 0; t < 2; ++t) {
      v16bf bf = load_frag(&Vs[(ntb + t) * 16 + lr][0], lane);
      acc[t] = wmma_bf16(af, bf, acc[t]);
    }
    __syncthreads();
  }

#pragma unroll
  for (int t = 0; t < 2; ++t) {
#pragma unroll
    for (int r = 0; r < 8; ++r) {
      const int d_ = mt * 16 + (lane >> 4) * 8 + r;
      const int e_ = (ntb + t) * 16 + lr;
      kvbT[((size_t)bh * 64 + e_) * 64 + d_] = (__bf16)acc[t][r];
    }
  }
}

// ---------------------------------------------------------------------------
// Kernel 6: x_update = (q @ kv) * si * alloc, stored bf16 as [B, N, H*D].
// Block: 128 tokens x 64 cols per (b,h); K=64 held entirely in LDS.
// The 64x64 bf16 kv tile is fetched by the Tensor Data Mover with the LDS
// pad feature reproducing the 72-element padded row stride:
//   row = 64 bf16 = 32 DWORDs  -> pad_interval code 4 (32 DWORDs)
//   pad = 8  bf16 =  4 DWORDs  -> pad_amount  code 3 (4 DWORDs)
// ---------------------------------------------------------------------------
__global__ __launch_bounds__(256) void xu_wmma_kernel(
    const float* __restrict__ q, const __bf16* __restrict__ kvbT,
    const float* __restrict__ mult, __bf16* __restrict__ xu) {
  const int bh = blockIdx.y, b = bh >> 3, h = bh & 7;
  const int n0 = blockIdx.x * 128;
  __shared__ __bf16 Qs[128][72];
  __shared__ __bf16 Bs[64][72];
  const int tid = threadIdx.x;
  const int lane = tid & 31, w = tid >> 5;
  const int lr = lane & 15;

  // ---- TDM: DMA the kv tile (64x64 bf16) into Bs, one descriptor, wave 0 ----
  if (w == 0) {
    const unsigned lds_base = (unsigned)(uintptr_t)(&Bs[0][0]);
    const unsigned long long ga =
        (unsigned long long)(uintptr_t)(kvbT + (size_t)bh * FA_D * FA_D);
    v4u g0;
    g0[0] = 1u;                                   // count=1, user descriptor
    g0[1] = lds_base;                             // lds_addr (bytes)
    g0[2] = (unsigned)(ga & 0xffffffffu);         // global_addr[31:0]
    g0[3] = (unsigned)((ga >> 32) & 0x01ffffffu)  // global_addr[56:32]
            | (2u << 30);                         // type = 2 ("image")
    v8i g1;
    g1[0] = (int)((1u << 16)      // data_size = 1 -> 2 bytes
                | (1u << 20)      // pad_enable
                | (4u << 22)      // pad_interval: 32 DWORDs
                | (3u << 25));    // pad_amount: 4 DWORDs
    g1[1] = (int)(64u << 16);     // tensor_dim0[15:0] = 64
    g1[2] = (int)(64u << 16);     // tensor_dim0 hi = 0 | tensor_dim1 lo = 64
    g1[3] = (int)(64u << 16);     // tensor_dim1 hi = 0 | tile_dim0 = 64
    g1[4] = 64;                   // tile_dim1 = 64 (tile_dim2 = 0)
    g1[5] = 64;                   // tensor_dim0_stride lo32 = 64
    g1[6] = 0;                    // stride0 hi | stride1 lo (unused for 2D)
    g1[7] = 0;
    v4i gz;
    gz[0] = 0; gz[1] = 0; gz[2] = 0; gz[3] = 0;   // groups 2/3 unused (2D)
    v8i g4z;
#pragma unroll
    for (int i = 0; i < 8; ++i) g4z[i] = 0;       // extra group (clang-23 form)
    __builtin_amdgcn_tensor_load_to_lds(g0, g1, gz, gz, g4z, 0);
    __builtin_amdgcn_s_wait_tensorcnt(0);
  }

  // ---- stage Q (fp32 -> bf16 conversion, overlaps the TDM transfer) ----
#pragma unroll
  for (int i = 0; i < 8; ++i) {   // 2048 float4 = 128x64 floats
    const int f = tid + i * 256;
    const int row = f >> 4;
    const int c4 = (f & 15) * 4;
    v4f qv = *(const v4f*)(q + ((size_t)bh * FA_N + n0 + row) * FA_D + c4);
#pragma unroll
    for (int j = 0; j < 4; ++j) Qs[row][c4 + j] = (__bf16)qv[j];
  }
  __syncthreads();

  v16bf a0 = load_frag(&Qs[w * 16 + lr][0], lane);
  v16bf a1 = load_frag(&Qs[w * 16 + lr][32], lane);
#pragma unroll
  for (int nt = 0; nt < 4; ++nt) {
    v16bf b0 = load_frag(&Bs[nt * 16 + lr][0], lane);
    v16bf b1 = load_frag(&Bs[nt * 16 + lr][32], lane);
    v8f acc = zero8();
    acc = wmma_bf16(a0, b0, acc);
    acc = wmma_bf16(a1, b1, acc);
#pragma unroll
    for (int r = 0; r < 8; ++r) {
      const int n = n0 + w * 16 + (lane >> 4) * 8 + r;
      const int e = nt * 16 + lr;
      const float val = acc[r] * mult[(size_t)bh * FA_N + n];
      xu[((size_t)b * FA_N + n) * FA_C + h * FA_D + e] = (__bf16)val;
    }
  }
}

// ---------------------------------------------------------------------------
// Kernel 7: out = x_update @ W_proj^T + b_proj  (M=32768, N=512, K=512)
// A tile (bf16 already) staged with GLOBAL_LOAD_ASYNC_TO_LDS_B128:
// 64x32 bf16 = 4096 B = 256 threads x 16 B, one async op per thread per step.
// W_proj staging (fp32->bf16 VALU) overlaps the in-flight async copy.
// ---------------------------------------------------------------------------
__global__ __launch_bounds__(256) void proj_wmma_kernel(
    const __bf16* __restrict__ xu, const float* __restrict__ Wp,
    const float* __restrict__ bp, float* __restrict__ out) {
  __shared__ __bf16 As[64][40];
  __shared__ __bf16 Bs[64][40];
  const int tid = threadIdx.x;
  const int lane = tid & 31, w = tid >> 5;
  const int lr = lane & 15;
  const int mt = w & 3, ntb = (w >> 2) * 2;
  const int m0 = blockIdx.x * 64;
  const int c0 = blockIdx.y * 64;

  const int arow = tid >> 2;          // 0..63
  const int aseg = (tid & 3) * 8;     // bf16 element offset: 0,8,16,24
  const unsigned lds_dst = (unsigned)(uintptr_t)(&As[arow][aseg]);

  v8f acc[2];
  acc[0] = zero8();
  acc[1] = zero8();

  for (int kk0 = 0; kk0 < FA_C; kk0 += 32) {
    // async DMA of the bf16 A tile straight into LDS (no VGPR round-trip)
    {
      const __bf16* gsrc = xu + (size_t)(m0 + arow) * FA_C + kk0 + aseg;
      asm volatile("global_load_async_to_lds_b128 %0, %1, off"
                   :: "v"(lds_dst), "v"(gsrc)
                   : "memory");
    }
#pragma unroll
    for (int i = 0; i < 2; ++i) {   // 512 float4 B elements (overlaps async)
      const int f = tid + i * 256;
      const int row = f >> 3;
      const int c4 = (f & 7) * 4;
      v4f wv = *(const v4f*)(Wp + (size_t)(c0 + row) * FA_C + kk0 + c4);
#pragma unroll
      for (int j = 0; j < 4; ++j) Bs[row][c4 + j] = (__bf16)wv[j];
    }
    asm volatile("s_wait_asynccnt 0x0" ::: "memory");
    __syncthreads();
    v16bf af = load_frag(&As[mt * 16 + lr][0], lane);
#pragma unroll
    for (int t = 0; t < 2; ++t) {
      v16bf bf = load_frag(&Bs[(ntb + t) * 16 + lr][0], lane);
      acc[t] = wmma_bf16(af, bf, acc[t]);
    }
    __syncthreads();
  }

#pragma unroll
  for (int t = 0; t < 2; ++t) {
#pragma unroll
    for (int r = 0; r < 8; ++r) {
      const int gm = m0 + mt * 16 + (lane >> 4) * 8 + r;
      const int gc = c0 + (ntb + t) * 16 + lr;
      out[(size_t)gm * FA_C + gc] = acc[t][r] + bp[gc];
    }
  }
}

// ---------------------------------------------------------------------------
// Launch
// ---------------------------------------------------------------------------
extern "C" void kernel_launch(void* const* d_in, const int* in_sizes, int n_in,
                              void* d_out, int out_size, void* d_ws, size_t ws_size,
                              hipStream_t stream) {
  (void)in_sizes; (void)n_in; (void)out_size; (void)ws_size;
  const float* x     = (const float*)d_in[0];
  const float* Wqkv  = (const float*)d_in[1];
  const float* Wproj = (const float*)d_in[2];
  const float* bproj = (const float*)d_in[3];
  float* out = (float*)d_out;

  // Workspace carving (fp32 region then bf16 region), all 16B-aligned.
  const size_t QKV = (size_t)FA_BH * FA_N * FA_D;  // 16,777,216
  const size_t TOK = (size_t)FA_BH * FA_N;         // 262,144
  float* q     = (float*)d_ws;
  float* k     = q + QKV;
  float* v     = k + QKV;
  float* si    = v + QKV;
  float* so    = si + TOK;
  float* mult  = so + TOK;
  float* expcs = mult + TOK;
  float* qsum  = expcs + TOK;
  float* ksum  = qsum + FA_BH * FA_D;
  float* ks_so = ksum + FA_BH * FA_D;
  float* qs_si = ks_so + FA_BH * FA_D;
  float* denom = qs_si + FA_BH * FA_D;
  __bf16* kvbT = (__bf16*)(denom + FA_BH);
  __bf16* xu   = kvbT + (size_t)FA_BH * FA_D * FA_D;

  // Zero atomic accumulators (capture-legal stream memsets).
  (void)hipMemsetAsync(ks_so, 0, 2 * FA_BH * FA_D * sizeof(float), stream);
  (void)hipMemsetAsync(denom, 0, FA_BH * sizeof(float), stream);

  qkv_wmma_kernel<<<dim3((FA_B * FA_N) / 64, (3 * FA_C) / 64), 256, 0, stream>>>(
      x, Wqkv, q, k, v);
  colsum_kernel<<<FA_BH, 256, 0, stream>>>(q, k, qsum, ksum);
  flow1_kernel<<<dim3(FA_N / 256, FA_BH), 256, 0, stream>>>(
      q, k, qsum, ksum, si, so, ks_so, qs_si);
  flow2_kernel<<<dim3(FA_N / 256, FA_BH), 256, 0, stream>>>(
      q, k, ks_so, qs_si, si, mult, expcs, denom);
  kv_wmma_kernel<<<FA_BH, 256, 0, stream>>>(k, v, expcs, denom, kvbT);
  xu_wmma_kernel<<<dim3(FA_N / 128, FA_BH), 256, 0, stream>>>(q, kvbT, mult, xu);
  proj_wmma_kernel<<<dim3((FA_B * FA_N) / 64, FA_C / 64), 256, 0, stream>>>(
      xu, Wproj, bproj, out);
}
